// LinearDeformableCrossAttnBlock_5231270166877
// MI455X (gfx1250) — matrix operand order repaired and checked by reading, not compiled, not run on hardware
//
#include <hip/hip_runtime.h>
#include <hip/hip_bf16.h>
#include <math.h>

// ---------------------------------------------------------------------------
// Types for CDNA5 WMMA (wave32): 16x16x32 bf16 -> f32
// ---------------------------------------------------------------------------
typedef __attribute__((ext_vector_type(16))) __bf16        v16bf;
typedef __attribute__((ext_vector_type(8)))  float         v8f;
typedef __attribute__((ext_vector_type(4)))  unsigned int  v4u;
typedef __attribute__((ext_vector_type(8)))  unsigned int  v8u;

// Problem constants (from reference setup_inputs)
#define BATCH   4
#define CCH     256
#define HH      64
#define WW      64
#define NTOK    4096          // H*W per batch
#define NHEADS  8
#define HDIM    32            // C / NHEADS
#define NPTS    4
#define MLP     1024

__device__ __forceinline__ unsigned short f2bf(float f) {
    unsigned int u = __float_as_uint(f);
    unsigned int r = u + 0x7FFFu + ((u >> 16) & 1u);   // round-to-nearest-even
    return (unsigned short)(r >> 16);
}

// ---------------------------------------------------------------------------
// Fragment loaders per CDNA5 ISA 7.12.2 layouts.
// A (16x32 bf16, MxK): lane L holds row M=L%16; with hi=L/16 the v16bf holds
//   K = {k0+8*hi .. +7} then {k0+16+8*hi .. +7}  (two contiguous 16B chunks)
// B (32x16 bf16, KxN): lane L holds column N=L%16; v16bf holds
//   K = {k0+8*hi .. +7} then {k0+16+8*hi .. +7}  (same as A). With W stored row-major
//   (out_features x in_features), B[k][n] == W[n][k], so this is W row n.
// ---------------------------------------------------------------------------
__device__ __forceinline__ v16bf load_a_frag(const unsigned short* __restrict__ Abase,
                                             int K, int k0, int m, int hi) {
    v4u lo = *(const v4u*)(Abase + (size_t)m * K + k0 + 8 * hi);
    v4u hc = *(const v4u*)(Abase + (size_t)m * K + k0 + 16 + 8 * hi);
    v8u t;
#pragma unroll
    for (int i = 0; i < 4; ++i) { t[i] = lo[i]; t[4 + i] = hc[i]; }
    return __builtin_bit_cast(v16bf, t);
}

__device__ __forceinline__ v16bf load_b_frag(const unsigned short* __restrict__ Wbase,
                                             int K, int k0, int n, int hi) {
    v4u lo = *(const v4u*)(Wbase + (size_t)n * K + k0 + 8 * hi);
    v4u hc = *(const v4u*)(Wbase + (size_t)n * K + k0 + 16 + 8 * hi);
    v8u t;
#pragma unroll
    for (int i = 0; i < 4; ++i) { t[i] = lo[i]; t[4 + i] = hc[i]; }
    return __builtin_bit_cast(v16bf, t);
}

// ---------------------------------------------------------------------------
// Generic WMMA GEMM: out[M,N] = A[M,K](bf16) @ W[N,K]^T(bf16) + bias, fused
// epilogues. One wave computes a 16x32 tile (2 accumulators share A frag).
// EPI 0: store f32 (ldN = N)                              [offset proj]
// EPI 1: + q_flat residual (strided src), store f32 x     [attn proj]
// EPI 2: exact GELU, store bf16                           [fc1]
// EPI 3: + x residual, store f32 transposed to (B,C,N)    [fc2 -> output]
// ---------------------------------------------------------------------------
template <int EPI>
__global__ __launch_bounds__(256)
void gemm_wmma_bf16(const unsigned short* __restrict__ A,
                    const unsigned short* __restrict__ Wt,
                    const float* __restrict__ bias,
                    float* __restrict__ outF,
                    unsigned short* __restrict__ outB,
                    const float* __restrict__ aux,
                    int M, int N, int K) {
    int wave = (blockIdx.x * blockDim.x + threadIdx.x) >> 5;
    int lane = threadIdx.x & 31;
    int tilesN2 = N >> 5;                  // 32-wide super tiles
    int tilesM  = M >> 4;
    if (wave >= tilesM * tilesN2) return;  // uniform per-wave: EXEC stays all-1s
    int tm = wave / tilesN2;
    int tn = (wave - tm * tilesN2) << 1;   // base 16-col tile index

    const unsigned short* Abase = A  + (size_t)tm * 16 * K;
    const unsigned short* W0    = Wt + (size_t)(tn + 0) * 16 * K;
    const unsigned short* W1    = Wt + (size_t)(tn + 1) * 16 * K;

    int m  = lane & 15;
    int hi = lane >> 4;

    v8f acc0 = {};
    v8f acc1 = {};
    for (int k0 = 0; k0 < K; k0 += 32) {
        v16bf a  = load_a_frag(Abase, K, k0, m, hi);
        v16bf b0 = load_b_frag(W0,    K, k0, m, hi);
        v16bf b1 = load_b_frag(W1,    K, k0, m, hi);
        acc0 = __builtin_amdgcn_wmma_f32_16x16x32_bf16(false, a, false, b0,
                                                       (short)0, acc0, false, false);
        acc1 = __builtin_amdgcn_wmma_f32_16x16x32_bf16(false, a, false, b1,
                                                       (short)0, acc1, false, false);
    }

    // C/D layout: VGPR r holds (M = r + 8*hi, N = lane%16)
    int colL = lane & 15;
#pragma unroll
    for (int r = 0; r < 8; ++r) {
        int row = tm * 16 + r + 8 * hi;
#pragma unroll
        for (int tt = 0; tt < 2; ++tt) {
            int   col = (tn + tt) * 16 + colL;
            float v   = (tt ? acc1[r] : acc0[r]) + bias[col];
            if (EPI == 0) {
                outF[(size_t)row * N + col] = v;
            } else if (EPI == 1) {
                int b = row >> 12, n = row & (NTOK - 1);
                v += aux[((size_t)b * CCH + col) * NTOK + n];   // q_flat residual
                outF[(size_t)row * CCH + col] = v;
            } else if (EPI == 2) {
                float g = 0.5f * v * (1.0f + erff(v * 0.70710678118654752f));
                outB[(size_t)row * N + col] = f2bf(g);
            } else {
                v += aux[(size_t)row * CCH + col];              // x residual
                int b = row >> 12, n = row & (NTOK - 1);
                outF[((size_t)b * CCH + col) * NTOK + n] = v;   // (B,C,H,W) store
            }
        }
    }
}

// ---------------------------------------------------------------------------
// LayerNorm over C=256 from channel-strided (B,C,N) source.
// 32 tokens per block via a conflict-free LDS transpose tile (coalesced loads).
// MODE 0: -> bf16 token-major (B*N, 256)            [qn for offset GEMM]
// MODE 1: -> f32 head-split  (B, Hd, D, N)          [kn for gather]
// ---------------------------------------------------------------------------
template <int MODE>
__global__ __launch_bounds__(256)
void ln_strided(const float* __restrict__ src,
                const float* __restrict__ g, const float* __restrict__ bta,
                unsigned short* __restrict__ outBf,
                float* __restrict__ outHead) {
    __shared__ float tile[32][257];
    __shared__ float ps[8][32], pss[8][32];
    __shared__ float mean_s[32], rstd_s[32];

    int t   = threadIdx.x;
    int blk = blockIdx.x;                  // B * (N/32) blocks
    int b   = blk >> 7;                    // 128 blocks per batch
    int n0  = (blk & 127) * 32;
    const float* base = src + (size_t)b * CCH * NTOK;

    int nl = t & 31, cg = t >> 5;          // cg in 0..7
#pragma unroll 4
    for (int i = 0; i < 32; ++i) {
        int c = i * 8 + cg;
        tile[nl][c] = base[(size_t)c * NTOK + n0 + nl];  // 128B coalesced
    }
    __syncthreads();

    {   // 8 partial reducers per token
        int tok = t & 31, part = t >> 5;
        float s = 0.f, ss = 0.f;
#pragma unroll 8
        for (int j = 0; j < 32; ++j) {
            float v = tile[tok][part * 32 + j];
            s += v; ss += v * v;
        }
        ps[part][tok] = s; pss[part][tok] = ss;
    }
    __syncthreads();
    if (t < 32) {
        float s = 0.f, ss = 0.f;
#pragma unroll
        for (int p = 0; p < 8; ++p) { s += ps[p][t]; ss += pss[p][t]; }
        float mu  = s * (1.0f / 256.0f);
        float var = ss * (1.0f / 256.0f) - mu * mu;
        mean_s[t] = mu;
        rstd_s[t] = rsqrtf(var + 1e-5f);
    }
    __syncthreads();

    if (MODE == 0) {
        float gg = g[t], bb = bta[t];
#pragma unroll 4
        for (int i = 0; i < 32; ++i) {
            float v = (tile[i][t] - mean_s[i]) * rstd_s[i] * gg + bb;
            outBf[((size_t)b * NTOK + n0 + i) * CCH + t] = f2bf(v);
        }
    } else {
        float mu = mean_s[nl], rs = rstd_s[nl];
#pragma unroll 4
        for (int cc = 0; cc < 32; ++cc) {
            int   c = cg * 32 + cc;
            float v = (tile[nl][c] - mu) * rs * g[c] + bta[c];
            // (B, Hd, D, N): head = cg, d = cc
            outHead[((size_t)(b * NHEADS + cg) * HDIM + cc) * NTOK + n0 + nl] = v;
        }
    }
}

// ---------------------------------------------------------------------------
// LayerNorm over contiguous 256-float rows -> bf16 (for fc1 input).
// ---------------------------------------------------------------------------
__global__ __launch_bounds__(256)
void ln_rows(const float* __restrict__ x,
             const float* __restrict__ g, const float* __restrict__ bta,
             unsigned short* __restrict__ outBf) {
    __shared__ float s1[256], s2[256];
    int row = blockIdx.x, t = threadIdx.x;
    float v = x[(size_t)row * CCH + t];
    s1[t] = v; s2[t] = v * v;
    __syncthreads();
#pragma unroll
    for (int off = 128; off > 0; off >>= 1) {
        if (t < off) { s1[t] += s1[t + off]; s2[t] += s2[t + off]; }
        __syncthreads();
    }
    float mu   = s1[0] * (1.0f / 256.0f);
    float var  = s2[0] * (1.0f / 256.0f) - mu * mu;
    float rstd = rsqrtf(var + 1e-5f);
    float o    = (v - mu) * rstd * g[t] + bta[t];
    outBf[(size_t)row * CCH + t] = f2bf(o);
}

// ---------------------------------------------------------------------------
// Deformable bilinear sampling + softmax point-weighting.
// Block = one (b, token); thread t: head = t/32, d = t%32.
// key slice per (b,head) is 512 KB -> L2-resident gathers.
// ---------------------------------------------------------------------------
__device__ __forceinline__ float samp_corner(const float* __restrict__ kp,
                                             float xf, float yf) {
    bool valid = (xf >= 0.f) && (xf <= 63.f) && (yf >= 0.f) && (yf <= 63.f);
    int xc = (int)fminf(fmaxf(xf, 0.f), 63.f);
    int yc = (int)fminf(fmaxf(yf, 0.f), 63.f);
    float v = kp[yc * WW + xc];
    return valid ? v : 0.f;
}

__global__ __launch_bounds__(256)
void deform_sample(const float* __restrict__ ow,    // (B*N, 96)
                   const float* __restrict__ kh,    // (B, Hd, D, N)
                   unsigned short* __restrict__ feat) { // (B*N, 256) bf16
    int bn = blockIdx.x;
    int t  = threadIdx.x;
    int head = t >> 5, d = t & 31;
    int b  = bn >> 12, n = bn & (NTOK - 1);
    int iy = n >> 6,   ix = n & 63;
    float gx = -1.0f + (2.0f / 63.0f) * (float)ix;
    float gy = -1.0f + (2.0f / 63.0f) * (float)iy;

    const float* orow = ow + (size_t)bn * 96 + head * 12;   // [p*3 + comp]
    float lg[4];
#pragma unroll
    for (int p = 0; p < 4; ++p) lg[p] = orow[p * 3 + 2];
    float mx = fmaxf(fmaxf(lg[0], lg[1]), fmaxf(lg[2], lg[3]));
    float e[4], s = 0.f;
#pragma unroll
    for (int p = 0; p < 4; ++p) { e[p] = expf(lg[p] - mx); s += e[p]; }
    float rs = 1.0f / s;

    const float* kp = kh + (((size_t)b * NHEADS + head) * HDIM + d) * NTOK;
    float acc = 0.f;
#pragma unroll
    for (int p = 0; p < 4; ++p) {
        float lx = gx + 0.1f * orow[p * 3 + 0];
        float ly = gy + 0.1f * orow[p * 3 + 1];
        float fx = (lx + 1.0f) * 0.5f * 63.0f;
        float fy = (ly + 1.0f) * 0.5f * 63.0f;
        float x0 = floorf(fx), y0 = floorf(fy);
        float wx1 = fx - x0, wy1 = fy - y0;
        float wx0 = 1.0f - wx1, wy0 = 1.0f - wy1;   // (x1 - x), (y1 - y)
        float bl = samp_corner(kp, x0,        y0       ) * wx0 * wy0
                 + samp_corner(kp, x0 + 1.0f, y0       ) * wx1 * wy0
                 + samp_corner(kp, x0,        y0 + 1.0f) * wx0 * wy1
                 + samp_corner(kp, x0 + 1.0f, y0 + 1.0f) * wx1 * wy1;
        acc += (e[p] * rs) * bl;
    }
    feat[(size_t)bn * CCH + t] = f2bf(acc);   // token-major, coalesced bf16
}

// ---------------------------------------------------------------------------
// fp32 -> bf16 weight conversion
// ---------------------------------------------------------------------------
__global__ void cvt_bf16(const float* __restrict__ src,
                         unsigned short* __restrict__ dst, int n) {
    int i = blockIdx.x * blockDim.x + threadIdx.x;
    if (i < n) dst[i] = f2bf(src[i]);
}

// ---------------------------------------------------------------------------
// Host orchestration
// ---------------------------------------------------------------------------
extern "C" void kernel_launch(void* const* d_in, const int* in_sizes, int n_in,
                              void* d_out, int out_size, void* d_ws, size_t ws_size,
                              hipStream_t stream) {
    (void)in_sizes; (void)n_in; (void)out_size; (void)ws_size;

    const float* query  = (const float*)d_in[0];
    const float* key    = (const float*)d_in[1];
    const float* n1g    = (const float*)d_in[2];
    const float* n1b    = (const float*)d_in[3];
    const float* off_w  = (const float*)d_in[4];
    const float* off_b  = (const float*)d_in[5];
    const float* proj_w = (const float*)d_in[6];
    const float* proj_b = (const float*)d_in[7];
    const float* n2g    = (const float*)d_in[8];
    const float* n2b    = (const float*)d_in[9];
    const float* fc1_w  = (const float*)d_in[10];
    const float* fc1_b  = (const float*)d_in[11];
    const float* fc2_w  = (const float*)d_in[12];
    const float* fc2_b  = (const float*)d_in[13];
    float* out = (float*)d_out;

    const int M = BATCH * NTOK;   // 16384 tokens

    char* ws = (char*)d_ws;
    size_t o = 0;
    auto alloc = [&](size_t bytes) {
        size_t cur = o;
        o += (bytes + 255) & ~(size_t)255;
        return cur;
    };
    unsigned short* qn    = (unsigned short*)(ws + alloc((size_t)M * CCH * 2));
    float*          kh    = (float*)(ws + alloc((size_t)BATCH * CCH * NTOK * 4));
    float*          owbuf = (float*)(ws + alloc((size_t)M * 96 * 4));
    unsigned short* feat  = (unsigned short*)(ws + alloc((size_t)M * CCH * 2));
    float*          x     = (float*)(ws + alloc((size_t)M * CCH * 4));
    unsigned short* xn    = (unsigned short*)(ws + alloc((size_t)M * CCH * 2));
    unsigned short* h     = (unsigned short*)(ws + alloc((size_t)M * MLP * 2));
    unsigned short* woff  = (unsigned short*)(ws + alloc((size_t)96 * CCH * 2));
    unsigned short* wproj = (unsigned short*)(ws + alloc((size_t)CCH * CCH * 2));
    unsigned short* wfc1  = (unsigned short*)(ws + alloc((size_t)MLP * CCH * 2));
    unsigned short* wfc2  = (unsigned short*)(ws + alloc((size_t)CCH * MLP * 2));

    // 1) weights -> bf16
    cvt_bf16<<<(96 * CCH + 255) / 256, 256, 0, stream>>>(off_w, woff, 96 * CCH);
    cvt_bf16<<<(CCH * CCH + 255) / 256, 256, 0, stream>>>(proj_w, wproj, CCH * CCH);
    cvt_bf16<<<(MLP * CCH + 255) / 256, 256, 0, stream>>>(fc1_w, wfc1, MLP * CCH);
    cvt_bf16<<<(CCH * MLP + 255) / 256, 256, 0, stream>>>(fc2_w, wfc2, CCH * MLP);

    // 2) layernorm q -> bf16 token-major ; layernorm k -> f32 head-split
    ln_strided<0><<<BATCH * (NTOK / 32), 256, 0, stream>>>(query, n1g, n1b, qn, nullptr);
    ln_strided<1><<<BATCH * (NTOK / 32), 256, 0, stream>>>(key,   n1g, n1b, nullptr, kh);

    // 3) offset projection: (M,96) = qn @ off_w^T
    {
        int waves = (M / 16) * (96 / 32);
        gemm_wmma_bf16<0><<<(waves * 32 + 255) / 256, 256, 0, stream>>>(
            qn, woff, off_b, owbuf, nullptr, nullptr, M, 96, CCH);
    }

    // 4) softmax + bilinear deformable gather -> feat bf16 (token-major)
    deform_sample<<<M, 256, 0, stream>>>(owbuf, kh, feat);

    // 5) output projection + q residual -> x f32
    {
        int waves = (M / 16) * (CCH / 32);
        gemm_wmma_bf16<1><<<(waves * 32 + 255) / 256, 256, 0, stream>>>(
            feat, wproj, proj_b, x, nullptr, query, M, CCH, CCH);
    }

    // 6) layernorm x -> xn bf16
    ln_rows<<<M, 256, 0, stream>>>(x, n2g, n2b, xn);

    // 7) fc1 + exact GELU -> h bf16
    {
        int waves = (M / 16) * (MLP / 32);
        gemm_wmma_bf16<2><<<(waves * 32 + 255) / 256, 256, 0, stream>>>(
            xn, wfc1, fc1_b, nullptr, h, nullptr, M, MLP, CCH);
    }

    // 8) fc2 + x residual, store transposed into (B,C,H,W) output
    {
        int waves = (M / 16) * (CCH / 32);
        gemm_wmma_bf16<3><<<(waves * 32 + 255) / 256, 256, 0, stream>>>(
            h, wfc2, fc2_b, out, nullptr, x, M, CCH, MLP);
    }
}
